// EntityModel_29643864277462
// MI455X (gfx1250) — matrix operand order repaired and checked by reading, not compiled, not run on hardware
//
#include <hip/hip_runtime.h>
#include <hip/hip_bf16.h>
#include <math.h>

// ---------------------------------------------------------------------------
// BiLSTM-CRF for MI455X (gfx1250, wave32, WMMA).
// GEMMs run on v_wmma_f32_16x16x32_f16 (f16 in / f32 accum).
// LDS staging uses the CDNA5 async copy path (global_load_async_to_lds_b128,
// tracked by ASYNCcnt) instead of a VGPR round-trip.
// ---------------------------------------------------------------------------

typedef _Float16 v16h __attribute__((ext_vector_type(16)));
typedef _Float16 v8h  __attribute__((ext_vector_type(8)));
typedef float    v8f  __attribute__((ext_vector_type(8)));

#define HID   512
#define EMB   768
#define NTAG  9
#define BQ    64
#define SQ    512
#define G4H   2048   // 4*HID

__device__ __forceinline__ v16h cat16(v8h lo, v8h hi) {
    return __builtin_shufflevector(lo, hi, 0,1,2,3,4,5,6,7,8,9,10,11,12,13,14,15);
}

__device__ __forceinline__ v8f wmma_f16(v16h a, v16h b, v8f c) {
    // D = A(16x32 f16) * B(32x16 f16) + C(16x16 f32)
    return __builtin_amdgcn_wmma_f32_16x16x32_f16(
        false, a, false, b, (short)0, c, false, false);
}

// Async copy of 16 bytes (per lane) global -> LDS.  VDST VGPR carries the LDS
// byte offset; for a generic pointer into the LDS aperture the low 32 bits are
// exactly that offset (ISA 10.2).  Tracked with ASYNCcnt.
__device__ __forceinline__ void async_g2lds_b128(void* lds_dst, const void* gsrc) {
    unsigned ldsoff = (unsigned)(unsigned long long)lds_dst;
    asm volatile("global_load_async_to_lds_b128 %0, %1, off"
                 :: "v"(ldsoff), "v"(gsrc)
                 : "memory");
}
__device__ __forceinline__ void wait_async0() {
    asm volatile("s_wait_asynccnt 0x0" ::: "memory");
}

// A fragment (16x32, f16) from row-major A[M][K]: per ISA 7.12.2,
// lane L holds row M=L&15; halves 0..7 -> K = k0 + (L>>4)*8 + h,
// halves 8..15 -> K = k0 + 16 + (L>>4)*8 + (h-8).  Two contiguous 16B runs.
__device__ __forceinline__ v16h load_afrag_g(const _Float16* __restrict__ A,
                                             int K, int row, int k0, int lane) {
    const _Float16* p = A + (size_t)row * K + k0 + ((lane >> 4) * 8);
    v8h lo = *(const v8h*)p;
    v8h hi = *(const v8h*)(p + 16);
    return cat16(lo, hi);
}

// B fragment (32x16, f16) from row-major Bn[N][K] (we compute A*B^T):
// lane L holds column N=L&15; halves 0..7 -> K = k0 + (L>>4)*8 + h,
// halves 8..15 -> K = k0 + 16 + (L>>4)*8 + (h-8).  Two contiguous 16B runs.
__device__ __forceinline__ v16h load_bfrag_g(const _Float16* __restrict__ Bn,
                                             int K, int nrow, int k0, int lane) {
    const _Float16* p = Bn + (size_t)nrow * K + k0 + ((lane >> 4) * 8);
    v8h lo = *(const v8h*)p;
    v8h hi = *(const v8h*)(p + 16);
    return cat16(lo, hi);
}

__device__ __forceinline__ float sigf(float x) { return 1.0f / (1.0f + __expf(-x)); }

// ---------------------------------------------------------------------------
// fp32 -> f16 conversion
// ---------------------------------------------------------------------------
__global__ void cvt_f32_f16(const float* __restrict__ in, _Float16* __restrict__ out,
                            size_t n) {
    size_t i = (size_t)blockIdx.x * blockDim.x + threadIdx.x;
    if (i < n) out[i] = (_Float16)in[i];
}

// ---------------------------------------------------------------------------
// zero init for recurrent state (runs every call: harness does not re-poison)
// ---------------------------------------------------------------------------
__global__ void init_state(float* __restrict__ c, _Float16* __restrict__ hp,
                           int nc, int nh) {
    int i = blockIdx.x * blockDim.x + threadIdx.x;
    if (i < nc) c[i] = 0.0f;
    if (i < nh) hp[i] = (_Float16)0.0f;
}

// ---------------------------------------------------------------------------
// C[M][N] = A[M][K] * B[N][K]^T + bias[N]    (A,B f16; C f32)
// block = 256 threads = 8 waves; wave w owns rows m0 = by*128 + w*16,
// computes a 16x64 strip (4 accumulators). B tile (64x32) staged in LDS via
// async global->LDS copies.  M%128==0, N%64==0, K%32==0.
// ---------------------------------------------------------------------------
__global__ __launch_bounds__(256) void gemm_bias_wmma(
    const _Float16* __restrict__ A, const _Float16* __restrict__ Bn,
    const float* __restrict__ bias, float* __restrict__ C,
    int M, int N, int K)
{
    __shared__ __align__(16) _Float16 bsh[64 * 32];

    const int tid  = threadIdx.x;
    const int lane = tid & 31;
    const int w    = tid >> 5;
    const int n0   = blockIdx.x * 64;
    const int m0   = blockIdx.y * 128 + w * 16;
    const int nl   = lane & 15;
    const int kr   = (lane >> 4) * 8;
    const int arow = m0 + nl;

    v8f acc0 = {}, acc1 = {}, acc2 = {}, acc3 = {};

    const int nload = tid >> 2;           // 0..63
    const int kload = (tid & 3) * 8;      // 0,8,16,24

    for (int k0 = 0; k0 < K; k0 += 32) {
        __syncthreads();   // previous-iteration consumers done before overwrite
        // async stage of B tile: bsh[n][k] = Bn[n0+n][k0+k]  (16B per thread)
        async_g2lds_b128(&bsh[nload * 32 + kload],
                         Bn + (size_t)(n0 + nload) * K + k0 + kload);
        if (k0 + 32 < K)
            __builtin_prefetch(Bn + (size_t)(n0 + nload) * K + k0 + 32 + kload, 0, 3);
        wait_async0();     // this wave's LDS writes have landed
        __syncthreads();   // publish to all waves

        v16h a = load_afrag_g(A, K, arow, k0, lane);

        const _Float16* bp0 = &bsh[(0 * 16 + nl) * 32 + kr];
        const _Float16* bp1 = &bsh[(1 * 16 + nl) * 32 + kr];
        const _Float16* bp2 = &bsh[(2 * 16 + nl) * 32 + kr];
        const _Float16* bp3 = &bsh[(3 * 16 + nl) * 32 + kr];
        v16h b0 = cat16(*(const v8h*)bp0, *(const v8h*)(bp0 + 16));
        v16h b1 = cat16(*(const v8h*)bp1, *(const v8h*)(bp1 + 16));
        v16h b2 = cat16(*(const v8h*)bp2, *(const v8h*)(bp2 + 16));
        v16h b3 = cat16(*(const v8h*)bp3, *(const v8h*)(bp3 + 16));

        acc0 = wmma_f16(a, b0, acc0);
        acc1 = wmma_f16(a, b1, acc1);
        acc2 = wmma_f16(a, b2, acc2);
        acc3 = wmma_f16(a, b3, acc3);
    }

    const float bi0 = bias[n0 + nl];
    const float bi1 = bias[n0 + 16 + nl];
    const float bi2 = bias[n0 + 32 + nl];
    const float bi3 = bias[n0 + 48 + nl];
    const int mbase = m0 + ((lane >> 4) * 8);   // C/D: M = r + 8*(lane>=16)
#pragma unroll
    for (int r = 0; r < 8; ++r) {
        size_t row = (size_t)(mbase + r) * N;
        C[row + n0 +      nl] = acc0[r] + bi0;
        C[row + n0 + 16 + nl] = acc1[r] + bi1;
        C[row + n0 + 32 + nl] = acc2[r] + bi2;
        C[row + n0 + 48 + nl] = acc3[r] + bi3;
    }
}

// ---------------------------------------------------------------------------
// One LSTM time step, both directions (blockIdx.z = dir).
// grid = (4 jn, 4 bm, 2 dir), 256 threads (8 waves).
// Wave owns hidden slice j0..j0+15 of 16 batch rows: 4 gate WMMA accumulators
// over K=HID, then fused sigmoid/tanh cell update (same lane slot across the
// 4 accumulators holds the same (b,j) element -> no divergence).
// h_prev is f16, double-buffered across launches to avoid cross-block races;
// the 16xHID h_prev tile is staged into LDS with async copies.
// ---------------------------------------------------------------------------
__global__ __launch_bounds__(256) void lstm_step(
    const _Float16* __restrict__ whh_f, const _Float16* __restrict__ whh_b,
    const float* __restrict__ xp_f, const float* __restrict__ xp_b,
    float* __restrict__ hf, float* __restrict__ hb,
    _Float16* __restrict__ hprev16,   // [dir][phase][B][HID] f16
    float* __restrict__ cstate,       // [dir][B][HID] f32
    int s)
{
    const int dir = blockIdx.z;
    const int t   = dir ? (SQ - 1 - s) : s;
    const _Float16* whh = dir ? whh_b : whh_f;
    const float*    xp  = dir ? xp_b  : xp_f;
    float*        hsout = dir ? hb    : hf;
    const size_t  BH    = (size_t)BQ * HID;
    const _Float16* hpin  = hprev16 + ((size_t)dir * 2 + (s & 1)) * BH;
    _Float16*       hpout = hprev16 + ((size_t)dir * 2 + ((s + 1) & 1)) * BH;
    float*          cst   = cstate + (size_t)dir * BH;

    const int tid  = threadIdx.x;
    const int lane = tid & 31;
    const int w    = tid >> 5;
    const int b0   = blockIdx.y * 16;
    const int j0   = (blockIdx.x * 8 + w) * 16;
    const int nl   = lane & 15;
    const int hk   = lane >> 4;

    __shared__ __align__(16) _Float16 hsh[16 * HID];   // 16 batch rows of h_prev

    {   // async stage: 32 contiguous halves (4 x 16B) per thread
        const int flat = tid * 32;
        const int row  = flat >> 9;      // /512
        const int k    = flat & 511;
        const _Float16* src = hpin + (size_t)(b0 + row) * HID + k;
        _Float16*       dst = &hsh[row * HID + k];
#pragma unroll
        for (int i = 0; i < 4; ++i)
            async_g2lds_b128(dst + i * 8, src + i * 8);
    }
    wait_async0();
    __syncthreads();

    v8f ai = {}, af = {}, ag = {}, ao = {};

    for (int k0 = 0; k0 < HID; k0 += 32) {
        const _Float16* ap = &hsh[nl * HID + k0 + hk * 8];
        v16h a = cat16(*(const v8h*)ap, *(const v8h*)(ap + 16));

        v16h bi = load_bfrag_g(whh, HID, 0 * HID + j0 + nl, k0, lane);
        v16h bf = load_bfrag_g(whh, HID, 1 * HID + j0 + nl, k0, lane);
        v16h bg = load_bfrag_g(whh, HID, 2 * HID + j0 + nl, k0, lane);
        v16h bo = load_bfrag_g(whh, HID, 3 * HID + j0 + nl, k0, lane);

        ai = wmma_f16(a, bi, ai);
        af = wmma_f16(a, bf, af);
        ag = wmma_f16(a, bg, ag);
        ao = wmma_f16(a, bo, ao);
    }

    const int j = j0 + nl;
#pragma unroll
    for (int r = 0; r < 8; ++r) {
        const int brow = b0 + r + hk * 8;
        const size_t xb = ((size_t)brow * SQ + t) * G4H;
        float pi = ai[r] + xp[xb +            j];
        float pf = af[r] + xp[xb +     HID  + j];
        float pg = ag[r] + xp[xb + 2 * HID  + j];
        float po = ao[r] + xp[xb + 3 * HID  + j];
        const size_t cidx = (size_t)brow * HID + j;
        float cn = sigf(pf) * cst[cidx] + sigf(pi) * tanhf(pg);
        float hn = sigf(po) * tanhf(cn);
        cst[cidx] = cn;
        hsout[((size_t)t * BQ + brow) * HID + j] = hn;
        hpout[cidx] = (_Float16)hn;
    }
}

// ---------------------------------------------------------------------------
// emis[s,b,t] = [hf(s,b,:) | hb(s,b,:)] . w_out[t,:] + b_out[t]
// one wave32 per (s,b) row; lane-parallel K, __shfl_xor reduction.
// ---------------------------------------------------------------------------
__global__ __launch_bounds__(256) void out_proj(
    const float* __restrict__ hf, const float* __restrict__ hb,
    const float* __restrict__ wout, const float* __restrict__ bout,
    float* __restrict__ emis)
{
    const int wid  = (blockIdx.x * 256 + threadIdx.x) >> 5;   // 0..S*B-1
    const int lane = threadIdx.x & 31;
    const float* pf = hf + (size_t)wid * HID;
    const float* pb = hb + (size_t)wid * HID;

    float acc[NTAG];
#pragma unroll
    for (int tg = 0; tg < NTAG; ++tg) acc[tg] = 0.0f;

    for (int k = lane; k < HID; k += 32) {
        float a = pf[k], b = pb[k];
#pragma unroll
        for (int tg = 0; tg < NTAG; ++tg)
            acc[tg] += a * wout[tg * (2 * HID) + k] + b * wout[tg * (2 * HID) + HID + k];
    }
#pragma unroll
    for (int off = 16; off > 0; off >>= 1)
#pragma unroll
        for (int tg = 0; tg < NTAG; ++tg)
            acc[tg] += __shfl_xor(acc[tg], off, 32);

    if (lane == 0)
#pragma unroll
        for (int tg = 0; tg < NTAG; ++tg)
            emis[(size_t)wid * NTAG + tg] = acc[tg] + bout[tg];
}

// ---------------------------------------------------------------------------
// CRF token-mean NLL. One block of 64 threads; thread b handles batch b.
// ---------------------------------------------------------------------------
__global__ __launch_bounds__(64) void crf_nll(
    const float* __restrict__ emis, const int* __restrict__ mask,
    const int* __restrict__ ttag,
    const float* __restrict__ start_tr, const float* __restrict__ end_tr,
    const float* __restrict__ trans, float* __restrict__ out)
{
    __shared__ float s_llh[BQ], s_m[BQ];
    __shared__ float tr[NTAG * NTAG], st[NTAG], et[NTAG];

    const int b = threadIdx.x;
    if (b < NTAG) { st[b] = start_tr[b]; et[b] = end_tr[b]; }
    for (int i = b; i < NTAG * NTAG; i += BQ) tr[i] = trans[i];
    __syncthreads();

    // tags[s,b] = ttag[b*S+s] ; m[s,b] = mask[b*S+s]
    int   prev = ttag[b * SQ + 0];
    float num  = st[prev] + emis[(size_t)(0 * BQ + b) * NTAG + prev];
    float alpha[NTAG];
#pragma unroll
    for (int t = 0; t < NTAG; ++t)
        alpha[t] = st[t] + emis[(size_t)(0 * BQ + b) * NTAG + t];
    float msum = (float)mask[b * SQ + 0];
    int   mtot = mask[b * SQ + 0];

    for (int s = 1; s < SQ; ++s) {
        const int   mi = mask[b * SQ + s];
        const float m  = (float)mi;
        mtot += mi;  msum += m;
        const int tg = ttag[b * SQ + s];
        const float* em = emis + ((size_t)s * BQ + b) * NTAG;
        num += (tr[prev * NTAG + tg] + em[tg]) * m;
        prev = tg;

        float nxt[NTAG];
#pragma unroll
        for (int tp = 0; tp < NTAG; ++tp) {
            float mx = -1e30f;
#pragma unroll
            for (int t = 0; t < NTAG; ++t)
                mx = fmaxf(mx, alpha[t] + tr[t * NTAG + tp]);
            float ssum = 0.0f;
#pragma unroll
            for (int t = 0; t < NTAG; ++t)
                ssum += __expf(alpha[t] + tr[t * NTAG + tp] - mx);
            nxt[tp] = mx + __logf(ssum) + em[tp];
        }
#pragma unroll
        for (int tp = 0; tp < NTAG; ++tp)
            alpha[tp] = (m > 0.0f) ? nxt[tp] : alpha[tp];
    }

    num += et[ttag[b * SQ + (mtot - 1)]];

    float mx = -1e30f;
#pragma unroll
    for (int t = 0; t < NTAG; ++t) mx = fmaxf(mx, alpha[t] + et[t]);
    float ssum = 0.0f;
#pragma unroll
    for (int t = 0; t < NTAG; ++t) ssum += __expf(alpha[t] + et[t] - mx);
    const float den = mx + __logf(ssum);

    s_llh[b] = num - den;
    s_m[b]   = msum;
    __syncthreads();
    if (b == 0) {
        float ls = 0.0f, ms = 0.0f;
        for (int i = 0; i < BQ; ++i) { ls += s_llh[i]; ms += s_m[i]; }
        out[0] = -(ls / ms);
    }
}

// ---------------------------------------------------------------------------
// host launcher
// ---------------------------------------------------------------------------
static inline size_t align256(size_t x) { return (x + 255) & ~(size_t)255; }

extern "C" void kernel_launch(void* const* d_in, const int* in_sizes, int n_in,
                              void* d_out, int out_size, void* d_ws, size_t ws_size,
                              hipStream_t stream) {
    (void)in_sizes; (void)n_in; (void)out_size; (void)ws_size;

    const float* x        = (const float*)d_in[0];
    const int*   mask     = (const int*)d_in[1];
    const int*   ttag     = (const int*)d_in[2];
    const float* w_ih_f   = (const float*)d_in[3];
    const float* w_hh_f   = (const float*)d_in[4];
    const float* b_f      = (const float*)d_in[5];
    const float* w_ih_b   = (const float*)d_in[6];
    const float* w_hh_b   = (const float*)d_in[7];
    const float* b_b      = (const float*)d_in[8];
    const float* w_out    = (const float*)d_in[9];
    const float* b_out    = (const float*)d_in[10];
    const float* start_tr = (const float*)d_in[11];
    const float* end_tr   = (const float*)d_in[12];
    const float* trans    = (const float*)d_in[13];

    char* ws = (char*)d_ws;
    size_t off = 0;
    _Float16* xf16   = (_Float16*)(ws + off); off = align256(off + (size_t)BQ*SQ*EMB*2);
    _Float16* wihf16 = (_Float16*)(ws + off); off = align256(off + (size_t)G4H*EMB*2);
    _Float16* wihb16 = (_Float16*)(ws + off); off = align256(off + (size_t)G4H*EMB*2);
    _Float16* whhf16 = (_Float16*)(ws + off); off = align256(off + (size_t)G4H*HID*2);
    _Float16* whhb16 = (_Float16*)(ws + off); off = align256(off + (size_t)G4H*HID*2);
    float*    xp_f   = (float*)(ws + off);    off = align256(off + (size_t)BQ*SQ*G4H*4);
    float*    xp_b   = (float*)(ws + off);    off = align256(off + (size_t)BQ*SQ*G4H*4);
    float*    hf     = (float*)(ws + off);    off = align256(off + (size_t)SQ*BQ*HID*4);
    float*    hb     = (float*)(ws + off);    off = align256(off + (size_t)SQ*BQ*HID*4);
    _Float16* hprev  = (_Float16*)(ws + off); off = align256(off + (size_t)2*2*BQ*HID*2);
    float*    cst    = (float*)(ws + off);    off = align256(off + (size_t)2*BQ*HID*4);
    float*    emis   = (float*)(ws + off);    off = align256(off + (size_t)SQ*BQ*NTAG*4);

    // 1) f32 -> f16 conversions
    {
        size_t n;
        n = (size_t)BQ*SQ*EMB;
        cvt_f32_f16<<<(unsigned)((n+255)/256), 256, 0, stream>>>(x, xf16, n);
        n = (size_t)G4H*EMB;
        cvt_f32_f16<<<(unsigned)((n+255)/256), 256, 0, stream>>>(w_ih_f, wihf16, n);
        cvt_f32_f16<<<(unsigned)((n+255)/256), 256, 0, stream>>>(w_ih_b, wihb16, n);
        n = (size_t)G4H*HID;
        cvt_f32_f16<<<(unsigned)((n+255)/256), 256, 0, stream>>>(w_hh_f, whhf16, n);
        cvt_f32_f16<<<(unsigned)((n+255)/256), 256, 0, stream>>>(w_hh_b, whhb16, n);
    }

    // 2) zero recurrent state (every call — deterministic)
    {
        int nc = 2 * BQ * HID;          // c state floats
        int nh = 2 * 2 * BQ * HID;      // h_prev halves (double-buffered)
        int mx = nh > nc ? nh : nc;
        init_state<<<(mx + 255) / 256, 256, 0, stream>>>(cst, hprev, nc, nh);
    }

    // 3) input projections: xp = x . W_ih^T + b   (M=B*S, N=4H, K=EMB)
    {
        dim3 grid(G4H / 64, (BQ * SQ) / 128);
        gemm_bias_wmma<<<grid, 256, 0, stream>>>(xf16, wihf16, b_f, xp_f,
                                                 BQ * SQ, G4H, EMB);
        gemm_bias_wmma<<<grid, 256, 0, stream>>>(xf16, wihb16, b_b, xp_b,
                                                 BQ * SQ, G4H, EMB);
    }

    // 4) recurrence: 512 sequential steps, both directions per launch
    {
        dim3 grid(4, 4, 2);
        for (int s = 0; s < SQ; ++s)
            lstm_step<<<grid, 256, 0, stream>>>(whhf16, whhb16, xp_f, xp_b,
                                                hf, hb, hprev, cst, s);
    }

    // 5) emissions
    out_proj<<<(SQ * BQ) / 8, 256, 0, stream>>>(hf, hb, w_out, b_out, emis);

    // 6) CRF NLL -> scalar
    crf_nll<<<1, BQ, 0, stream>>>(emis, mask, ttag, start_tr, end_tr, trans,
                                  (float*)d_out);
}
